// VectorQuantizer_30648886624708
// MI455X (gfx1250) — matrix operand order repaired and checked by reading, not compiled, not run on hardware
//
#include <hip/hip_runtime.h>
#include <hip/hip_bf16.h>

// ---------------------------------------------------------------------------
// Sinkhorn vector quantizer for MI455X (gfx1250, wave32).
//  - d-matrix (32768x512 fp32, 64MB) computed once via split-f16 WMMA
//    (v_wmma_f32_16x16x32_f16 x3 per K-chunk => ~fp32 GEMM accuracy),
//    kept L2-resident (64MB < 192MB L2) for 100 Sinkhorn sweeps.
//  - Sinkhorn done in log2 domain with only column scalings g_j (row
//    scalings cancel in argmax): one fused row-softmax pass per iteration,
//    deterministic fp64 cross-wave column reduction (no float atomics).
// Workspace requirement: ~70 MB.
// ---------------------------------------------------------------------------

#define N_ROWSQ  32768              // 16*2048 latent vectors
#define NE       512
#define EDIM     64
#define NCOL     512
#define SK_ITERS 100
#define SK_WAVES 1024
#define SK_ROWS  (N_ROWSQ / SK_WAVES)   // 32 rows per wave
#define AO_WAVES 4096
#define AO_ROWS  (N_ROWSQ / AO_WAVES)   // 8 rows per wave
#define GEMM_BLOCKS 512                 // 4 waves/block, 1 row-tile/wave
#define XQ_ELEMS (N_ROWSQ * EDIM)       // 2097152
#define LOSS_OFF XQ_ELEMS
#define IDX_OFF  (XQ_ELEMS + 1)
#define L2E 1.4426950408889634f

typedef __attribute__((ext_vector_type(16))) _Float16 v16h;
typedef __attribute__((ext_vector_type(8)))  float    v8f;
typedef __attribute__((ext_vector_type(4)))  float    v4f;
typedef __attribute__((ext_vector_type(2)))  float    v2f;

__device__ inline float fexp2(float x){
#if __has_builtin(__builtin_amdgcn_exp2f)
  return __builtin_amdgcn_exp2f(x);          // v_exp_f32 (base-2)
#else
  return exp2f(x);
#endif
}
__device__ inline float flog2(float x){
#if __has_builtin(__builtin_amdgcn_logf)
  return __builtin_amdgcn_logf(x);           // v_log_f32 (base-2)
#else
  return log2f(x);
#endif
}

// --------------------------------------------------------------------------
// ||x_i||^2 and ||c_j||^2 : one wave per row of 64 floats.
// --------------------------------------------------------------------------
__global__ __launch_bounds__(256) void k_norms(const float* __restrict__ x,
                                               const float* __restrict__ cb,
                                               float* __restrict__ nx,
                                               float* __restrict__ nc){
  int gw   = (int)((blockIdx.x * blockDim.x + threadIdx.x) >> 5);
  int lane = (int)(threadIdx.x & 31);
  const float* p;
  if (gw < N_ROWSQ)            p = x  + (size_t)gw * EDIM;
  else if (gw < N_ROWSQ + NE)  p = cb + (size_t)(gw - N_ROWSQ) * EDIM;
  else return;
  v2f v = *(const v2f*)(p + 2 * lane);
  float s = v[0]*v[0] + v[1]*v[1];
#pragma unroll
  for (int off = 16; off > 0; off >>= 1) s += __shfl_xor(s, off);
  if (lane == 0){
    if (gw < N_ROWSQ) nx[gw] = s; else nc[gw - N_ROWSQ] = s;
  }
}

// Convert 8 consecutive floats into (hi, residual) f16 halves of a v16h pair.
__device__ inline void cvt8(const float* p, v16h& h, v16h& l, int base){
  v4f a = *(const v4f*)p;
  v4f b = *(const v4f*)(p + 4);
#pragma unroll
  for (int e = 0; e < 4; ++e){
    _Float16 ha = (_Float16)a[e]; h[base + e]     = ha; l[base + e]     = (_Float16)(a[e] - (float)ha);
    _Float16 hb = (_Float16)b[e]; h[base + 4 + e] = hb; l[base + 4 + e] = (_Float16)(b[e] - (float)hb);
  }
}

// --------------------------------------------------------------------------
// d_ij = ||x_i||^2 + ||c_j||^2 - 2 x_i.c_j via split-f16 WMMA.
// One wave per 16-row tile, looping over all 32 column tiles (A reused).
// Also tracks per-block min/max of d for the centering step.
// --------------------------------------------------------------------------
__global__ __launch_bounds__(128) void k_gemm(const float* __restrict__ x,
                                              const float* __restrict__ cb,
                                              const float* __restrict__ nx,
                                              const float* __restrict__ nc,
                                              float* __restrict__ dmat,
                                              float* __restrict__ mmpart){
  int wave = (int)(threadIdx.x >> 5), lane = (int)(threadIdx.x & 31);
  int rt   = (int)blockIdx.x * 4 + wave;          // row tile 0..2047
  int lrow = lane & 15, hi = lane >> 4;

  // A-matrix (16x32 f16) layout: lanes 0-15 K-base 0, lanes 16-31 K-base 8;
  // elems 0..7 -> K = kb..kb+7, elems 8..15 -> K = 16+kb..16+kb+7.
  const float* xr = x + (size_t)(rt * 16 + lrow) * EDIM;
  int kbase = hi * 8;
  v16h ah[2], al[2];
#pragma unroll
  for (int kk = 0; kk < 2; ++kk){
    cvt8(xr + 32 * kk + kbase,      ah[kk], al[kk], 0);
    cvt8(xr + 32 * kk + 16 + kbase, ah[kk], al[kk], 8);
  }
  float nxv[8];
#pragma unroll
  for (int r = 0; r < 8; ++r) nxv[r] = nx[rt * 16 + hi * 8 + r];

  float mn = 3.0e38f, mx = -3.0e38f;
  int koff = hi * 8;                              // B-matrix K-base 0/8, same as A
  for (int ct = 0; ct < 32; ++ct){
    int n = ct * 16 + lrow;                       // codebook entry (B column)
    const float* cr = cb + (size_t)n * EDIM;
    v8f acc = {0.f,0.f,0.f,0.f,0.f,0.f,0.f,0.f};
#pragma unroll
    for (int kk = 0; kk < 2; ++kk){
      v16h bh, bl;
      cvt8(cr + 32 * kk + koff,     bh, bl, 0);
      cvt8(cr + 32 * kk + koff + 16, bh, bl, 8);
      acc = __builtin_amdgcn_wmma_f32_16x16x32_f16(false, ah[kk], false, bh, (short)0, acc, false, false);
      acc = __builtin_amdgcn_wmma_f32_16x16x32_f16(false, al[kk], false, bh, (short)0, acc, false, false);
      acc = __builtin_amdgcn_wmma_f32_16x16x32_f16(false, ah[kk], false, bl, (short)0, acc, false, false);
    }
    float ncv = nc[n];
#pragma unroll
    for (int r = 0; r < 8; ++r){                  // D: m = hi*8 + r, col = lrow
      float dv = nxv[r] + ncv - 2.0f * acc[r];
      dmat[(size_t)(rt * 16 + hi * 8 + r) * NCOL + n] = dv;
      mn = fminf(mn, dv); mx = fmaxf(mx, dv);
    }
  }
#pragma unroll
  for (int off = 16; off > 0; off >>= 1){
    mn = fminf(mn, __shfl_xor(mn, off));
    mx = fmaxf(mx, __shfl_xor(mx, off));
  }
  __shared__ float smn[4], smx[4];
  if (lane == 0){ smn[wave] = mn; smx[wave] = mx; }
  __syncthreads();
  if (threadIdx.x == 0){
    float bmn = smn[0], bmx = smx[0];
    for (int w = 1; w < 4; ++w){ bmn = fminf(bmn, smn[w]); bmx = fmaxf(bmx, smx[w]); }
    mmpart[2 * blockIdx.x]     = bmn;
    mmpart[2 * blockIdx.x + 1] = bmx;
  }
}

// --------------------------------------------------------------------------
// Reduce per-block min/max -> centering scalars, init g2 = 0.
// scal[0] = s2 = log2(e)/(eps*amp) ; scal[1] = middle*s2
// (so t_ij = scal[1] - d_ij*scal[0] is log2 of the Gibbs kernel)
// --------------------------------------------------------------------------
__global__ __launch_bounds__(512) void k_prep(const float* __restrict__ mmpart,
                                              float* __restrict__ scal,
                                              float* __restrict__ g2){
  int t = (int)threadIdx.x;
  float mn = mmpart[2 * t], mx = mmpart[2 * t + 1];
#pragma unroll
  for (int off = 16; off > 0; off >>= 1){
    mn = fminf(mn, __shfl_xor(mn, off));
    mx = fmaxf(mx, __shfl_xor(mx, off));
  }
  __shared__ float smn[16], smx[16];
  if ((t & 31) == 0){ smn[t >> 5] = mn; smx[t >> 5] = mx; }
  __syncthreads();
  if (t == 0){
    for (int w = 1; w < 16; ++w){ mn = fminf(mn, smn[w]); mx = fmaxf(mx, smx[w]); }
    float middle = 0.5f * (mx + mn);
    float amp    = mx - middle + 1e-5f;
    float s2     = L2E / (0.003f * amp);
    scal[0] = s2; scal[1] = middle * s2;
  }
  g2[t] = 0.0f;
}

// --------------------------------------------------------------------------
// One Sinkhorn iteration, row stage: per row i compute log2-sum-exp over j
// of (t_ij + g_j), accumulate row-softmax into per-wave column partials.
// Each wave owns 32 rows and all 512 columns (16 per lane, j = 4*lane+128*k+c).
// --------------------------------------------------------------------------
__global__ __launch_bounds__(256) void k_sk_row(const float* __restrict__ dmat,
                                                const float* __restrict__ g2,
                                                const float* __restrict__ scal,
                                                float* __restrict__ colpart){
  int gw   = (int)((blockIdx.x * blockDim.x + threadIdx.x) >> 5);
  int lane = (int)(threadIdx.x & 31);
  float s2 = scal[0], msc = scal[1];
  float o[16], acc[16];
#pragma unroll
  for (int k = 0; k < 4; ++k){
    v4f g = *(const v4f*)(g2 + 4 * lane + 128 * k);
#pragma unroll
    for (int c = 0; c < 4; ++c){ o[4*k+c] = msc + g[c]; acc[4*k+c] = 0.f; }
  }
  const float* base = dmat + (size_t)gw * SK_ROWS * NCOL;
  for (int r = 0; r < SK_ROWS; ++r){
    const float* row = base + (size_t)r * NCOL;
    float tv[16];
#pragma unroll
    for (int k = 0; k < 4; ++k){
      v4f d = *(const v4f*)(row + 4 * lane + 128 * k);
#pragma unroll
      for (int c = 0; c < 4; ++c) tv[4*k+c] = fmaf(d[c], -s2, o[4*k+c]);
    }
    float mxv = tv[0];
#pragma unroll
    for (int e = 1; e < 16; ++e) mxv = fmaxf(mxv, tv[e]);
#pragma unroll
    for (int off = 16; off > 0; off >>= 1) mxv = fmaxf(mxv, __shfl_xor(mxv, off));
    float ex[16]; float sum = 0.f;
#pragma unroll
    for (int e = 0; e < 16; ++e){ ex[e] = fexp2(tv[e] - mxv); sum += ex[e]; }
#pragma unroll
    for (int off = 16; off > 0; off >>= 1) sum += __shfl_xor(sum, off);
    float rinv = 1.0f / sum;
#pragma unroll
    for (int e = 0; e < 16; ++e) acc[e] = fmaf(ex[e], rinv, acc[e]);
  }
  // transposed partials: colpart[j][wave] -> coalesced deterministic reduce
#pragma unroll
  for (int k = 0; k < 4; ++k)
#pragma unroll
    for (int c = 0; c < 4; ++c)
      colpart[(size_t)(4 * lane + 128 * k + c) * SK_WAVES + gw] = acc[4*k+c];
}

// --------------------------------------------------------------------------
// Column stage: S_j = sum over 1024 wave partials (fp64, fixed order),
// g_j += log2(N/K) - log2(S_j)   with log2(N/K) = 6 exactly.
// One wave per column.
// --------------------------------------------------------------------------
__global__ __launch_bounds__(256) void k_sk_update(const float* __restrict__ colpart,
                                                   float* __restrict__ g2){
  int j    = (int)((blockIdx.x * blockDim.x + threadIdx.x) >> 5);
  int lane = (int)(threadIdx.x & 31);
  const float* p = colpart + (size_t)j * SK_WAVES;
  double s = 0.0;
  for (int t = 0; t < SK_WAVES / 32; ++t) s += (double)p[lane + 32 * t];
#pragma unroll
  for (int off = 16; off > 0; off >>= 1) s += __shfl_xor(s, off);
  if (lane == 0) g2[j] = g2[j] + 6.0f - flog2((float)s);
}

// --------------------------------------------------------------------------
// Final pass: argmax_j (t_ij + g_j) (first index on ties, like jnp.argmax),
// gather codebook row -> x_q_st, per-wave loss partials, indices as floats.
// --------------------------------------------------------------------------
__global__ __launch_bounds__(256) void k_out(const float* __restrict__ dmat,
                                             const float* __restrict__ g2,
                                             const float* __restrict__ scal,
                                             const float* __restrict__ x,
                                             const float* __restrict__ cb,
                                             float* __restrict__ out,
                                             float* __restrict__ losspart){
  int gw   = (int)((blockIdx.x * blockDim.x + threadIdx.x) >> 5);
  int lane = (int)(threadIdx.x & 31);
  float s2 = scal[0], msc = scal[1];
  float o[16];
#pragma unroll
  for (int k = 0; k < 4; ++k){
    v4f g = *(const v4f*)(g2 + 4 * lane + 128 * k);
#pragma unroll
    for (int c = 0; c < 4; ++c) o[4*k+c] = msc + g[c];
  }
  float lsum = 0.f;
  for (int r = 0; r < AO_ROWS; ++r){
    int i = gw * AO_ROWS + r;
    const float* row = dmat + (size_t)i * NCOL;
    float bv = -3.0e38f; int bj = 1 << 30;
#pragma unroll
    for (int k = 0; k < 4; ++k){
      v4f d = *(const v4f*)(row + 4 * lane + 128 * k);
#pragma unroll
      for (int c = 0; c < 4; ++c){          // j ascends within lane -> first-tie kept
        float t = fmaf(d[c], -s2, o[4*k+c]);
        int j = 4 * lane + 128 * k + c;
        if (t > bv){ bv = t; bj = j; }
      }
    }
#pragma unroll
    for (int off = 16; off > 0; off >>= 1){
      float ov = __shfl_xor(bv, off);
      int   oj = __shfl_xor(bj, off);
      if (ov > bv || (ov == bv && oj < bj)){ bv = ov; bj = oj; }
    }
    if (lane == 0) out[IDX_OFF + i] = (float)bj;
    const float* cq = cb + (size_t)bj * EDIM;
    v2f q  = *(const v2f*)(cq + 2 * lane);
    v2f xv = *(const v2f*)(x + (size_t)i * EDIM + 2 * lane);
    *(v2f*)(out + (size_t)i * EDIM + 2 * lane) = q;   // x_q_st == x_q in value
    float dx = q[0] - xv[0], dy = q[1] - xv[1];
    lsum += dx * dx + dy * dy;
  }
#pragma unroll
  for (int off = 16; off > 0; off >>= 1) lsum += __shfl_xor(lsum, off);
  if (lane == 0) losspart[gw] = lsum;
}

// loss = (1 + BETA) * mean((x_q - x)^2), deterministic fp64 tree reduce.
__global__ __launch_bounds__(256) void k_loss(const float* __restrict__ losspart,
                                              float* __restrict__ out){
  __shared__ double sd[256];
  int t = (int)threadIdx.x;
  double s = 0.0;
  for (int u = 0; u < AO_WAVES / 256; ++u) s += (double)losspart[t + 256 * u];
  sd[t] = s; __syncthreads();
  for (int off = 128; off > 0; off >>= 1){
    if (t < off) sd[t] += sd[t + off];
    __syncthreads();
  }
  if (t == 0) out[LOSS_OFF] = (float)(1.25 * sd[0] / (double)XQ_ELEMS);
}

extern "C" void kernel_launch(void* const* d_in, const int* in_sizes, int n_in,
                              void* d_out, int out_size, void* d_ws, size_t ws_size,
                              hipStream_t stream){
  (void)in_sizes; (void)n_in; (void)out_size; (void)ws_size;
  const float* x  = (const float*)d_in[0];
  const float* cb = (const float*)d_in[1];
  float* out = (float*)d_out;

  float* ws       = (float*)d_ws;
  float* dmat     = ws;                                   // 32768*512 fp32 (64 MB)
  float* nx       = dmat + (size_t)N_ROWSQ * NCOL;        // 32768
  float* ncn      = nx + N_ROWSQ;                         // 512
  float* g2       = ncn + NE;                             // 512 (16B aligned)
  float* scal     = g2 + NE;                              // 16
  float* mmpart   = scal + 16;                            // 2*512
  float* colpart  = mmpart + 2 * GEMM_BLOCKS;             // 512*1024 (2 MB)
  float* losspart = colpart + (size_t)NCOL * SK_WAVES;    // 4096

  k_norms<<<(N_ROWSQ + NE) / 8, 256, 0, stream>>>(x, cb, nx, ncn);
  k_gemm <<<GEMM_BLOCKS, 128, 0, stream>>>(x, cb, nx, ncn, dmat, mmpart);
  k_prep <<<1, 512, 0, stream>>>(mmpart, scal, g2);
  for (int it = 0; it < SK_ITERS; ++it){
    k_sk_row   <<<SK_WAVES / 8, 256, 0, stream>>>(dmat, g2, scal, colpart);
    k_sk_update<<<NCOL / 8,    256, 0, stream>>>(colpart, g2);
  }
  k_out <<<AO_WAVES / 8, 256, 0, stream>>>(dmat, g2, scal, x, cb, out, losspart);
  k_loss<<<1, 256, 0, stream>>>(losspart, out);
}
